// ObjGraph_50929722196324
// MI455X (gfx1250) — compile-verified, matrix operand order checked by reading, not run on hardware
//
#include <hip/hip_runtime.h>
#include <hip/hip_bf16.h>

typedef __attribute__((ext_vector_type(16))) _Float16 v16h;
typedef __attribute__((ext_vector_type(8)))  _Float16 v8h;
typedef __attribute__((ext_vector_type(8)))  float    v8f;

#define BG   16
#define NGn  4096
#define Nn   65536
#define En   1048576
#define E2n  1114112          /* E + N self loops */
#define Hn   64
#define Kn   3277
#define NKn  52432            /* BG * Kn */
#define FNEG (-3.402823466e38f)

__device__ __forceinline__ void atomicMaxF(float* addr, float v) {
  if (v >= 0.f) atomicMax((int*)addr, __float_as_int(v));
  else          atomicMin((unsigned int*)addr, __float_as_uint(v));
}

// ---------------------------------------------------------------- utility
__global__ void fill_f32(float* p, float v, int n) {
  int i = blockIdx.x * blockDim.x + threadIdx.x;
  if (i < n) p[i] = v;
}

__global__ void f32_to_f16(const float* __restrict__ s, _Float16* __restrict__ d, int n) {
  int i = blockIdx.x * blockDim.x + threadIdx.x;
  if (i < n) d[i] = (_Float16)s[i];
}

// ------------------------------------------------- edge scatter (conv agg)
// msg: agg[dst] += h[src] * wmul ;  deg[dst] += vdeg
__global__ void scatter_conv_kernel(const int* __restrict__ src, const int* __restrict__ dst,
                                    const float* __restrict__ wmul, const float* __restrict__ vdeg,
                                    const float* __restrict__ h,
                                    float* __restrict__ agg, float* __restrict__ deg, int nE) {
  int t = blockIdx.x * blockDim.x + threadIdx.x;
  if (t >= nE * 16) return;
  int e  = t >> 4;
  int c4 = (t & 15) << 2;
  int s = src[e], d = dst[e];
  float wm = wmul ? wmul[e] : 1.f;
  float4 v = *(const float4*)(h + (size_t)s * Hn + c4);
  float* ap = agg + (size_t)d * Hn + c4;
  atomicAdd(ap + 0, v.x * wm);
  atomicAdd(ap + 1, v.y * wm);
  atomicAdd(ap + 2, v.z * wm);
  atomicAdd(ap + 3, v.w * wm);
  if ((t & 15) == 0) atomicAdd(deg + d, vdeg ? vdeg[e] : 1.f);
}

// --------------------------------------------------------- WMMA GEMM core
// Fragment = two contiguous 8-half runs (ISA 16-bit A layout):
//   elems 0..7  -> base_ptr[kk+kb .. kk+kb+7]
//   elems 8..15 -> base_ptr[kk+kb+16 .. kk+kb+23]
__device__ __forceinline__ v16h frag_ld(const _Float16* p, int kk, int kb) {
  v8h lo = *(const v8h*)(p + kk + kb);
  v8h hi = *(const v8h*)(p + kk + kb + 16);
  return __builtin_shufflevector(lo, hi, 0, 1, 2, 3, 4, 5, 6, 7,
                                 8, 9, 10, 11, 12, 13, 14, 15);
}

// out = act( (agg / max(deg,1)) @ W1 + bias [+ x2 @ W2] ), [n,64]x[64,64]
// 256 threads = 8 waves; block covers 64 rows; wave -> col tile (wv&3),
// row tiles ((wv>>2)*2 + {0,1}).
__global__ void conv_out_kernel(const float* __restrict__ agg,
                                const float* __restrict__ deg,   // nullable
                                const float* __restrict__ x2,    // nullable
                                const _Float16* __restrict__ W1, // [64][64] k-major
                                const _Float16* __restrict__ W2, // nullable
                                const float* __restrict__ bias,
                                float* __restrict__ out, int do_relu, int nrows) {
  __shared__ _Float16 A1[64 * Hn];
  __shared__ _Float16 A2[64 * Hn];
  __shared__ _Float16 W1t[Hn * Hn];   // transposed: [n][k]
  __shared__ _Float16 W2t[Hn * Hn];
  const int tid  = threadIdx.x;          // 256
  const int row0 = blockIdx.x * 64;

  for (int idx = tid; idx < Hn * Hn; idx += 256) {
    int k = idx >> 6, n = idx & 63;
    W1t[n * Hn + k] = W1[idx];
  }
  if (x2) for (int idx = tid; idx < Hn * Hn; idx += 256) {
    int k = idx >> 6, n = idx & 63;
    W2t[n * Hn + k] = W2[idx];
  }
  for (int idx = tid; idx < 64 * Hn; idx += 256) {
    int r = idx >> 6, c = idx & 63;
    int row = row0 + r; if (row >= nrows) row = nrows - 1;   // clamp (last block)
    size_t gi = (size_t)row * Hn + c;
    float v = agg[gi];
    if (deg) v /= fmaxf(deg[row], 1.f);
    A1[idx] = (_Float16)v;
    if (x2) A2[idx] = (_Float16)x2[gi];
  }
  __syncthreads();

  const int lane = tid & 31;
  const int wv   = tid >> 5;             // 0..7
  const int m    = lane & 15;
  const int kb   = (lane >> 4) * 8;
  const int col  = (wv & 3) * 16 + m;    // output / B column
  const float bv = bias[col];
  const _Float16* w1p = W1t + col * Hn;
  const _Float16* w2p = W2t + col * Hn;

  #pragma unroll
  for (int rt = 0; rt < 2; ++rt) {
    const int rowTile = ((wv >> 2) * 2 + rt) * 16;
    const _Float16* a1p = A1 + (rowTile + m) * Hn;
    const _Float16* a2p = A2 + (rowTile + m) * Hn;

    v8f acc;
    #pragma unroll
    for (int r = 0; r < 8; ++r) acc[r] = bv;

    #pragma unroll
    for (int kk = 0; kk < 64; kk += 32) {
      v16h a = frag_ld(a1p, kk, kb);
      v16h b = frag_ld(w1p, kk, kb);
      acc = __builtin_amdgcn_wmma_f32_16x16x32_f16(false, a, false, b,
                                                   (short)0, acc, false, false);
    }
    if (x2) {
      #pragma unroll
      for (int kk = 0; kk < 64; kk += 32) {
        v16h a = frag_ld(a2p, kk, kb);
        v16h b = frag_ld(w2p, kk, kb);
        acc = __builtin_amdgcn_wmma_f32_16x16x32_f16(false, a, false, b,
                                                     (short)0, acc, false, false);
      }
    }
    #pragma unroll
    for (int r = 0; r < 8; ++r) {
      int row = row0 + rowTile + r + ((lane >> 4) << 3);
      if (row < nrows) {
        float v = acc[r];
        if (do_relu) v = fmaxf(v, 0.f);
        out[(size_t)row * Hn + col] = v;
      }
    }
  }
}

// -------------------------------------------------------- global mean pool
__global__ void gmean_kernel(const float* __restrict__ X, float* __restrict__ Z,
                             int rpg, int colOff, float inv) {
  int g = blockIdx.x, c = threadIdx.x;   // 64 threads
  const float* p = X + (size_t)g * rpg * Hn + c;
  float s = 0.f;
  for (int r = 0; r < rpg; ++r) s += p[(size_t)r * Hn];
  Z[g * 256 + colOff + c] = s * inv;
}

// ----------------------------------------------------------- ASAP kernels
__device__ __forceinline__ void edge2(int e, const int* src, const int* dst, int& s, int& d) {
  if (e < En) { s = src[e]; d = dst[e]; } else { s = d = e - En; }
}

__global__ void xq_max_kernel(const int* __restrict__ src, const int* __restrict__ dst,
                              const float* __restrict__ h, float* __restrict__ xq) {
  int t = blockIdx.x * blockDim.x + threadIdx.x;
  if (t >= E2n * 16) return;
  int e = t >> 4, c4 = (t & 15) << 2, s, d;
  edge2(e, src, dst, s, d);
  float4 v = *(const float4*)(h + (size_t)s * Hn + c4);
  float* qp = xq + (size_t)d * Hn + c4;
  atomicMaxF(qp + 0, v.x); atomicMaxF(qp + 1, v.y);
  atomicMaxF(qp + 2, v.z); atomicMaxF(qp + 3, v.w);
}

__global__ void att_dots_kernel(const float* __restrict__ xq, const float* __restrict__ h,
                                const float* __restrict__ att_w,
                                float* __restrict__ qd, float* __restrict__ hd) {
  int i = blockIdx.x * blockDim.x + threadIdx.x;
  if (i >= Nn) return;
  float a = 0.f, b = 0.f;
  #pragma unroll
  for (int k = 0; k < Hn; ++k) {
    a += xq[(size_t)i * Hn + k] * att_w[k];
    b += h [(size_t)i * Hn + k] * att_w[Hn + k];
  }
  qd[i] = a; hd[i] = b;
}

__global__ void score_max_kernel(const int* src, const int* dst,
                                 const float* __restrict__ qd, const float* __restrict__ hd,
                                 const float* __restrict__ att_b,
                                 float* __restrict__ sc, float* __restrict__ mx) {
  int e = blockIdx.x * blockDim.x + threadIdx.x;
  if (e >= E2n) return;
  int s, d; edge2(e, src, dst, s, d);
  float v = qd[d] + hd[s] + att_b[0];
  v = (v > 0.f) ? v : 0.2f * v;            // leaky_relu 0.2
  sc[e] = v;
  atomicMaxF(mx + d, v);
}

__global__ void exp_sum_kernel(const int* src, const int* dst,
                               float* __restrict__ sc, const float* __restrict__ mx,
                               float* __restrict__ den) {
  int e = blockIdx.x * blockDim.x + threadIdx.x;
  if (e >= E2n) return;
  int s, d; edge2(e, src, dst, s, d); (void)s;
  float x = expf(sc[e] - mx[d]);
  sc[e] = x;
  atomicAdd(den + d, x);
}

__global__ void norm_kernel(const int* src, const int* dst,
                            float* __restrict__ sc, const float* __restrict__ den) {
  int e = blockIdx.x * blockDim.x + threadIdx.x;
  if (e >= E2n) return;
  int s, d; edge2(e, src, dst, s, d); (void)s;
  sc[e] = sc[e] / fmaxf(den[d], 1e-16f);
}

__global__ void xn_scatter_kernel(const int* src, const int* dst,
                                  const float* __restrict__ h, const float* __restrict__ sc,
                                  float* __restrict__ xn) {
  int t = blockIdx.x * blockDim.x + threadIdx.x;
  if (t >= E2n * 16) return;
  int e = t >> 4, c4 = (t & 15) << 2, s, d;
  edge2(e, src, dst, s, d);
  float w = sc[e];
  float4 v = *(const float4*)(h + (size_t)s * Hn + c4);
  float* p = xn + (size_t)d * Hn + c4;
  atomicAdd(p + 0, v.x * w); atomicAdd(p + 1, v.y * w);
  atomicAdd(p + 2, v.z * w); atomicAdd(p + 3, v.w * w);
}

__global__ void le_dots_kernel(const float* __restrict__ xn,
                               const float* __restrict__ w1, const float* __restrict__ b1,
                               const float* __restrict__ w2, const float* __restrict__ w3,
                               float* __restrict__ aa, float* __restrict__ bb,
                               float* __restrict__ cc) {
  int i = blockIdx.x * blockDim.x + threadIdx.x;
  if (i >= Nn) return;
  float a = 0.f, b = 0.f, c = 0.f;
  #pragma unroll
  for (int k = 0; k < Hn; ++k) {
    float x = xn[(size_t)i * Hn + k];
    a += x * w1[k]; b += x * w2[k]; c += x * w3[k];
  }
  aa[i] = a + b1[0]; bb[i] = b; cc[i] = c;
}

__global__ void fit_scatter_kernel(const int* src, const int* dst,
                                   const float* __restrict__ bb,
                                   float* __restrict__ sbb, float* __restrict__ deg2) {
  int e = blockIdx.x * blockDim.x + threadIdx.x;
  if (e >= E2n) return;
  int s, d; edge2(e, src, dst, s, d);
  atomicAdd(sbb + d, bb[s]);
  atomicAdd(deg2 + d, 1.f);
}

__global__ void fitness_kernel(const float* __restrict__ aa, const float* __restrict__ deg2,
                               const float* __restrict__ sbb, const float* __restrict__ cc,
                               const float* __restrict__ le3_b, float* __restrict__ fit) {
  int i = blockIdx.x * blockDim.x + threadIdx.x;
  if (i >= Nn) return;
  float f = aa[i] * deg2[i] - sbb[i] + cc[i] + le3_b[0];
  fit[i] = 1.f / (1.f + expf(-f));
}

// per-graph top-K via binary search on float bit patterns (all in (0,1))
__global__ void topk_kernel(const float* __restrict__ fit,
                            int* __restrict__ o2n, int* __restrict__ n2o) {
  __shared__ unsigned ub[NGn];
  __shared__ int red[256];
  __shared__ int ctr_gt, ctr_eq;
  int g = blockIdx.x, tid = threadIdx.x;
  for (int i = tid; i < NGn; i += 256) ub[i] = __float_as_uint(fit[g * NGn + i]);
  if (tid == 0) { ctr_gt = 0; ctr_eq = 0; }
  __syncthreads();

  unsigned lo = 0u, hi = 0x3F800001u;      // count_ge(lo)>=K, count_ge(hi)<K
  while (hi - lo > 1u) {
    unsigned mid = lo + (hi - lo) / 2u;
    int c = 0;
    for (int i = tid; i < NGn; i += 256) c += (ub[i] >= mid);
    red[tid] = c; __syncthreads();
    for (int s = 128; s > 0; s >>= 1) { if (tid < s) red[tid] += red[tid + s]; __syncthreads(); }
    int tot = red[0]; __syncthreads();
    if (tot >= Kn) lo = mid; else hi = mid;
  }
  unsigned T = lo;
  int c = 0;
  for (int i = tid; i < NGn; i += 256) c += (ub[i] > T);
  red[tid] = c; __syncthreads();
  for (int s = 128; s > 0; s >>= 1) { if (tid < s) red[tid] += red[tid + s]; __syncthreads(); }
  int cgt = red[0];
  int extra = Kn - cgt;
  __syncthreads();

  for (int i = tid; i < NGn; i += 256) {
    unsigned u = ub[i];
    int old = g * NGn + i, nid = -1;
    if (u > T) {
      int p = atomicAdd(&ctr_gt, 1);
      nid = g * Kn + p;
    } else if (u == T) {
      int p = atomicAdd(&ctr_eq, 1);
      if (p < extra) nid = g * Kn + cgt + p;
    }
    o2n[old] = nid;
    if (nid >= 0) n2o[nid] = old;
  }
}

__global__ void pool_x_kernel(const float* __restrict__ xn, const float* __restrict__ fit,
                              const int* __restrict__ n2o, float* __restrict__ xo) {
  int t = blockIdx.x * blockDim.x + threadIdx.x;
  if (t >= NKn * Hn) return;
  int j = t >> 6, c = t & 63;
  int old = n2o[j];
  xo[t] = xn[(size_t)old * Hn + c] * fit[old];
}

__global__ void remap_kernel(const int* src, const int* dst, const int* __restrict__ o2n,
                             float* __restrict__ sc /*in-place -> new weight*/,
                             float* __restrict__ vn,
                             int* __restrict__ so, int* __restrict__ do_) {
  int e = blockIdx.x * blockDim.x + threadIdx.x;
  if (e >= E2n) return;
  int s, d; edge2(e, src, dst, s, d);
  int nr = o2n[s], nc = o2n[d];
  int kept = (nr >= 0) && (nc >= 0);
  sc[e] = kept ? sc[e] : 0.f;
  vn[e] = kept ? 1.f : 0.f;
  so[e] = nr >= 0 ? nr : 0;
  do_[e] = nc >= 0 ? nc : 0;
}

// ------------------------------------------------------------- final MLP
__global__ void mlp_kernel(const float* __restrict__ Z,
                           const float* __restrict__ W1, const float* __restrict__ b1,
                           const float* __restrict__ W2, const float* __restrict__ b2,
                           float* __restrict__ out) {
  __shared__ float t1[16 * 64];
  __shared__ float t2[16 * 10];
  int tid = threadIdx.x;                       // 256 threads, 1 block
  for (int idx = tid; idx < 16 * 64; idx += 256) {
    int r = idx >> 6, c = idx & 63;
    float s = b1[c];
    for (int k = 0; k < 256; ++k) s += Z[r * 256 + k] * W1[k * 64 + c];
    t1[idx] = fmaxf(s, 0.f);
  }
  __syncthreads();
  for (int idx = tid; idx < 160; idx += 256) {
    int r = idx / 10, c = idx % 10;
    float s = b2[c];
    for (int k = 0; k < 64; ++k) s += t1[r * 64 + k] * W2[k * 10 + c];
    t2[idx] = s;
  }
  __syncthreads();
  if (tid < 16) {
    float m = FNEG;
    for (int c = 0; c < 10; ++c) m = fmaxf(m, t2[tid * 10 + c]);
    float se = 0.f;
    for (int c = 0; c < 10; ++c) se += expf(t2[tid * 10 + c] - m);
    float lse = m + logf(se);
    for (int c = 0; c < 10; ++c) out[tid * 10 + c] = t2[tid * 10 + c] - lse;
  }
}

// ---------------------------------------------------------------- driver
extern "C" void kernel_launch(void* const* d_in, const int* in_sizes, int n_in,
                              void* d_out, int out_size, void* d_ws, size_t ws_size,
                              hipStream_t stream) {
  const float* x       = (const float*)d_in[0];
  const int*   eidx    = (const int*)  d_in[1];
  const int*   esrc    = eidx;
  const int*   edst    = eidx + En;
  const float* W1r     = (const float*)d_in[3];
  const float* W1t     = (const float*)d_in[4];
  const float* b1      = (const float*)d_in[5];
  const float* Wcr     = (const float*)d_in[6];
  const float* Wct     = (const float*)d_in[7];
  const float* bc      = (const float*)d_in[8];
  const float* pl_W    = (const float*)d_in[9];
  const float* pl_b    = (const float*)d_in[10];
  const float* att_w   = (const float*)d_in[11];
  const float* att_b   = (const float*)d_in[12];
  const float* le1_w   = (const float*)d_in[13];
  const float* le1_b   = (const float*)d_in[14];
  const float* le2_w   = (const float*)d_in[15];
  const float* le3_w   = (const float*)d_in[16];
  const float* le3_b   = (const float*)d_in[17];
  const float* lin1_W  = (const float*)d_in[18];
  const float* lin1_b  = (const float*)d_in[19];
  const float* lin2_W  = (const float*)d_in[20];
  const float* lin2_b  = (const float*)d_in[21];

  char* w = (char*)d_ws;
  size_t off = 0;
  auto alloc = [&](size_t bytes) -> void* {
    void* p = w + off;
    off = (off + bytes + 255) & ~(size_t)255;
    return p;
  };
  const size_t N64 = (size_t)Nn * Hn;
  float* A    = (float*)alloc(N64 * 4);            // agg / xq / pooled agg
  float* H1   = (float*)alloc(N64 * 4);            // h1 -> xn -> h5
  float* H2   = (float*)alloc(N64 * 4);            // h2 -> h4
  float* XOUT = (float*)alloc((size_t)NKn * Hn * 4);
  float* DEG  = (float*)alloc(Nn * 4);             // deg / deg2
  float* QD   = (float*)alloc(Nn * 4);
  float* HD   = (float*)alloc(Nn * 4);
  float* AA   = (float*)alloc(Nn * 4);
  float* BB   = (float*)alloc(Nn * 4);
  float* CC   = (float*)alloc(Nn * 4);
  float* FIT  = (float*)alloc(Nn * 4);
  float* MX   = (float*)alloc(Nn * 4);
  float* DEN  = (float*)alloc(Nn * 4);
  float* SBB  = (float*)alloc(Nn * 4);
  float* SC   = (float*)alloc((size_t)E2n * 4);    // score -> edge weight
  float* VN   = (float*)alloc((size_t)E2n * 4);
  int*   SRCO = (int*)  alloc((size_t)E2n * 4);
  int*   DSTO = (int*)  alloc((size_t)E2n * 4);
  int*   O2N  = (int*)  alloc(Nn * 4);
  int*   N2O  = (int*)  alloc(NKn * 4);
  float* Z    = (float*)alloc(BG * 256 * 4);
  _Float16* WH = (_Float16*)alloc(9 * 4096 * 2);
  _Float16* W1r_h = WH;
  _Float16* W1t_h = WH + 4096;
  _Float16* Wcr_h = WH + 8192;    // 3 matrices
  _Float16* Wct_h = WH + 20480;   // 3 matrices
  _Float16* plW_h = WH + 32768;
  (void)in_sizes; (void)n_in; (void)out_size; (void)ws_size;

  const int TB = 256;
  auto gs = [](long long n, int tb) { return (int)((n + tb - 1) / tb); };

  // weight conversion to f16
  f32_to_f16<<<gs(4096, TB), TB, 0, stream>>>(W1r, W1r_h, 4096);
  f32_to_f16<<<gs(4096, TB), TB, 0, stream>>>(W1t, W1t_h, 4096);
  f32_to_f16<<<gs(12288, TB), TB, 0, stream>>>(Wcr, Wcr_h, 12288);
  f32_to_f16<<<gs(12288, TB), TB, 0, stream>>>(Wct, Wct_h, 12288);
  f32_to_f16<<<gs(4096, TB), TB, 0, stream>>>(pl_W, plW_h, 4096);

  // ---- conv1: x -> H1
  fill_f32<<<gs(N64, TB), TB, 0, stream>>>(A, 0.f, (int)N64);
  fill_f32<<<gs(Nn, TB), TB, 0, stream>>>(DEG, 0.f, Nn);
  scatter_conv_kernel<<<gs((long long)En * 16, TB), TB, 0, stream>>>(
      esrc, edst, nullptr, nullptr, x, A, DEG, En);
  conv_out_kernel<<<gs(Nn, 64), 256, 0, stream>>>(A, DEG, x, W1r_h, W1t_h, b1, H1, 1, Nn);
  gmean_kernel<<<BG, 64, 0, stream>>>(H1, Z, NGn, 0, 1.f / NGn);

  // ---- conv2 (i=0): H1 -> H2
  fill_f32<<<gs(N64, TB), TB, 0, stream>>>(A, 0.f, (int)N64);
  fill_f32<<<gs(Nn, TB), TB, 0, stream>>>(DEG, 0.f, Nn);
  scatter_conv_kernel<<<gs((long long)En * 16, TB), TB, 0, stream>>>(
      esrc, edst, nullptr, nullptr, H1, A, DEG, En);
  conv_out_kernel<<<gs(Nn, 64), 256, 0, stream>>>(A, DEG, H1, Wcr_h, Wct_h, bc, H2, 1, Nn);
  gmean_kernel<<<BG, 64, 0, stream>>>(H2, Z, NGn, 64, 1.f / NGn);

  // ---- ASAPooling on H2
  fill_f32<<<gs(N64, TB), TB, 0, stream>>>(A, FNEG, (int)N64);
  xq_max_kernel<<<gs((long long)E2n * 16, TB), TB, 0, stream>>>(esrc, edst, H2, A);
  conv_out_kernel<<<gs(Nn, 64), 256, 0, stream>>>(A, nullptr, nullptr, plW_h, nullptr,
                                                  pl_b, A, 0, Nn);     // xq in-place
  att_dots_kernel<<<gs(Nn, TB), TB, 0, stream>>>(A, H2, att_w, QD, HD);
  fill_f32<<<gs(Nn, TB), TB, 0, stream>>>(MX, FNEG, Nn);
  score_max_kernel<<<gs(E2n, TB), TB, 0, stream>>>(esrc, edst, QD, HD, att_b, SC, MX);
  fill_f32<<<gs(Nn, TB), TB, 0, stream>>>(DEN, 0.f, Nn);
  exp_sum_kernel<<<gs(E2n, TB), TB, 0, stream>>>(esrc, edst, SC, MX, DEN);
  norm_kernel<<<gs(E2n, TB), TB, 0, stream>>>(esrc, edst, SC, DEN);
  fill_f32<<<gs(N64, TB), TB, 0, stream>>>(H1, 0.f, (int)N64);         // H1 = xn
  xn_scatter_kernel<<<gs((long long)E2n * 16, TB), TB, 0, stream>>>(esrc, edst, H2, SC, H1);
  le_dots_kernel<<<gs(Nn, TB), TB, 0, stream>>>(H1, le1_w, le1_b, le2_w, le3_w, AA, BB, CC);
  fill_f32<<<gs(Nn, TB), TB, 0, stream>>>(SBB, 0.f, Nn);
  fill_f32<<<gs(Nn, TB), TB, 0, stream>>>(DEG, 0.f, Nn);               // deg2
  fit_scatter_kernel<<<gs(E2n, TB), TB, 0, stream>>>(esrc, edst, BB, SBB, DEG);
  fitness_kernel<<<gs(Nn, TB), TB, 0, stream>>>(AA, DEG, SBB, CC, le3_b, FIT);
  topk_kernel<<<BG, 256, 0, stream>>>(FIT, O2N, N2O);
  pool_x_kernel<<<gs((long long)NKn * Hn, TB), TB, 0, stream>>>(H1, FIT, N2O, XOUT);
  remap_kernel<<<gs(E2n, TB), TB, 0, stream>>>(esrc, edst, O2N, SC, VN, SRCO, DSTO);

  // ---- conv3 (i=1): XOUT -> H2(reused)
  fill_f32<<<gs((long long)NKn * Hn, TB), TB, 0, stream>>>(A, 0.f, NKn * Hn);
  fill_f32<<<gs(NKn, TB), TB, 0, stream>>>(DEG, 0.f, NKn);
  scatter_conv_kernel<<<gs((long long)E2n * 16, TB), TB, 0, stream>>>(
      SRCO, DSTO, SC, VN, XOUT, A, DEG, E2n);
  conv_out_kernel<<<gs(NKn, 64), 256, 0, stream>>>(A, DEG, XOUT, Wcr_h + 4096, Wct_h + 4096,
                                                   bc + Hn, H2, 1, NKn);
  gmean_kernel<<<BG, 64, 0, stream>>>(H2, Z, Kn, 128, 1.f / Kn);

  // ---- conv4 (i=2): H2 -> H1(reused)
  fill_f32<<<gs((long long)NKn * Hn, TB), TB, 0, stream>>>(A, 0.f, NKn * Hn);
  fill_f32<<<gs(NKn, TB), TB, 0, stream>>>(DEG, 0.f, NKn);
  scatter_conv_kernel<<<gs((long long)E2n * 16, TB), TB, 0, stream>>>(
      SRCO, DSTO, SC, VN, H2, A, DEG, E2n);
  conv_out_kernel<<<gs(NKn, 64), 256, 0, stream>>>(A, DEG, H2, Wcr_h + 8192, Wct_h + 8192,
                                                   bc + 2 * Hn, H1, 1, NKn);
  gmean_kernel<<<BG, 64, 0, stream>>>(H1, Z, Kn, 192, 1.f / Kn);

  // ---- MLP head + log_softmax
  mlp_kernel<<<1, 256, 0, stream>>>(Z, lin1_W, lin1_b, lin2_W, lin2_b, (float*)d_out);
}
